// Seq2Seq_24558622998991
// MI455X (gfx1250) — matrix operand order repaired and checked by reading, not compiled, not run on hardware
//
#include <hip/hip_runtime.h>
#include <math.h>

// ---------------------------------------------------------------------------
// Types for WMMA fragments (wave32, gfx1250)
// ---------------------------------------------------------------------------
typedef __attribute__((ext_vector_type(8)))  float  v8f;
typedef __attribute__((ext_vector_type(8)))  __bf16 v8bf;
typedef __attribute__((ext_vector_type(16))) __bf16 v16bf;

#define HID 128            // H == E == 128
#define G3H 384            // 3*H
#define VOCAB 32000
#define BATCH 64
#define SRC_T 64
#define DEC_T 63           // T-1
#define M_ROWS (BATCH * DEC_T)   // 4032
#define M_GROUPS (M_ROWS / 32)   // 126  (two 16-row tiles per wave)
#define N_GROUPS (VOCAB / 64)    // 500

__device__ __forceinline__ float sigmoidf_(float x) {
    return 1.0f / (1.0f + __expf(-x));
}

// ---------------------------------------------------------------------------
// fc_W f32 -> bf16 conversion (one-time, 4.1M elements)
// ---------------------------------------------------------------------------
__global__ __launch_bounds__(256) void cvt_bf16_kernel(
    const float* __restrict__ in, __bf16* __restrict__ out, int n)
{
    int i = blockIdx.x * 256 + threadIdx.x;
    if (i < n) out[i] = (__bf16)in[i];
}

// ---------------------------------------------------------------------------
// Precompute input-side gates: gi[row][j] = bih[j] + emb[tok] . Wih[j]
// ---------------------------------------------------------------------------
__global__ __launch_bounds__(G3H) void gi_kernel(
    const int*   __restrict__ tok,
    int steps, int tok_stride,
    const float* __restrict__ emb,   // [V][128]
    const float* __restrict__ Wih,   // [384][128]
    const float* __restrict__ bih,   // [384]
    float*       __restrict__ gi)    // [B*steps][384]
{
    __shared__ float x[HID];
    const int row = blockIdx.x;
    const int j   = threadIdx.x;
    const int b   = row / steps;
    const int t   = row - b * steps;
    const int token = tok[b * tok_stride + t];
    if (j < HID) x[j] = emb[(size_t)token * HID + j];
    __syncthreads();
    const float* w = Wih + (size_t)j * HID;
    float acc = bih[j];
#pragma unroll 8
    for (int k = 0; k < HID; ++k) acc += x[k] * w[k];
    gi[(size_t)row * G3H + j] = acc;
}

// ---------------------------------------------------------------------------
// GRU recurrence: one block per batch row, 128 threads (one per h element).
// n = tanh(i_n + r * (h@Whh_n + bhh_n)); h' = (1-z)*n + z*h   (PyTorch conv.)
// ---------------------------------------------------------------------------
__global__ __launch_bounds__(HID) void gru_scan_kernel(
    const float* __restrict__ gi,    // [B*steps][384]
    const float* __restrict__ h0,    // [B][128] or nullptr => zeros
    const float* __restrict__ Whh,   // [384][128]
    const float* __restrict__ bhh,   // [384]
    float*       __restrict__ hT,    // [B][128] or nullptr
    __bf16*      __restrict__ outs,  // [B*steps][128] or nullptr
    int steps)
{
    __shared__ float h[HID];
    const int b = blockIdx.x;
    const int j = threadIdx.x;

    h[j] = h0 ? h0[b * HID + j] : 0.0f;
    const float* wr = Whh + (size_t)(        j) * HID;
    const float* wz = Whh + (size_t)(HID   + j) * HID;
    const float* wn = Whh + (size_t)(2*HID + j) * HID;
    const float br = bhh[j], bz = bhh[HID + j], bn = bhh[2*HID + j];
    __syncthreads();

    for (int t = 0; t < steps; ++t) {
        const float* g = gi + ((size_t)b * steps + t) * G3H;
        float ar = 0.0f, az = 0.0f, an = 0.0f;
#pragma unroll 4
        for (int k = 0; k < HID; ++k) {
            const float hk = h[k];
            ar += wr[k] * hk;
            az += wz[k] * hk;
            an += wn[k] * hk;
        }
        const float r  = sigmoidf_(g[j]         + ar + br);
        const float z  = sigmoidf_(g[HID + j]   + az + bz);
        const float nn = tanhf    (g[2*HID + j] + r * (an + bn));
        const float hj = h[j];
        const float hnew = (1.0f - z) * nn + z * hj;
        __syncthreads();
        h[j] = hnew;
        if (outs) outs[((size_t)b * steps + t) * HID + j] = (__bf16)hnew;
        __syncthreads();
    }
    if (hT) hT[b * HID + j] = h[j];
}

// ---------------------------------------------------------------------------
// FC GEMM: logits[m][n] = sum_k A[m][k] * W[n][k] + bias[n]
// Each wave: 32(M) x 64(N) = 8 f32 accumulators; two A fragments; K=128 in
// 4 bf16 WMMA steps of 8 WMMAs each. Each B fragment feeds 2 WMMAs,
// halving L2 traffic for W and amortizing load waits.
// ---------------------------------------------------------------------------
__global__ __launch_bounds__(128) void fc_gemm_kernel(
    const __bf16* __restrict__ A,
    const __bf16* __restrict__ W,
    const float*  __restrict__ bias,
    float*        __restrict__ out)
{
    const int lane  = threadIdx.x & 31;
    const int wave  = blockIdx.x * 4 + (threadIdx.x >> 5);
    const int mgrp  = wave % M_GROUPS;       // 0..125  -> rows mgrp*32
    const int ngrp  = wave / M_GROUPS;       // 0..499  -> cols ngrp*64
    const int hi16  = lane >> 4;             // 0 | 1
    const int l15   = lane & 15;

    // A (16x32 bf16): lane holds row l15; K = hi16*16 .. +15
    const int kb_a = hi16 * 16;
    const __bf16* arow0 = A + (size_t)(mgrp * 32 + l15     ) * HID + kb_a;
    const __bf16* arow1 = A + (size_t)(mgrp * 32 + l15 + 16) * HID + kb_a;

    // B (32x16 bf16): lane holds column n = fc_W row; K = hi16*16 .. +15
    const int n0   = ngrp * 64 + l15;
    const int kb_b = hi16 * 16;
    const __bf16* wrow0 = W + (size_t)(n0     ) * HID + kb_b;
    const __bf16* wrow1 = W + (size_t)(n0 + 16) * HID + kb_b;
    const __bf16* wrow2 = W + (size_t)(n0 + 32) * HID + kb_b;
    const __bf16* wrow3 = W + (size_t)(n0 + 48) * HID + kb_b;

    v8f acc[2][4] = {};

#pragma unroll
    for (int kk = 0; kk < HID; kk += 32) {
        // --- issue ALL loads for this K-step first ---
        v8bf a0l = *(const v8bf*)(arow0 + kk);
        v8bf a0h = *(const v8bf*)(arow0 + kk + 8);
        v8bf a1l = *(const v8bf*)(arow1 + kk);
        v8bf a1h = *(const v8bf*)(arow1 + kk + 8);
        v8bf b0l = *(const v8bf*)(wrow0 + kk);
        v8bf b0h = *(const v8bf*)(wrow0 + kk + 8);
        v8bf b1l = *(const v8bf*)(wrow1 + kk);
        v8bf b1h = *(const v8bf*)(wrow1 + kk + 8);
        v8bf b2l = *(const v8bf*)(wrow2 + kk);
        v8bf b2h = *(const v8bf*)(wrow2 + kk + 8);
        v8bf b3l = *(const v8bf*)(wrow3 + kk);
        v8bf b3h = *(const v8bf*)(wrow3 + kk + 8);

        v16bf af0 = __builtin_shufflevector(a0l, a0h,
            0,1,2,3,4,5,6,7,8,9,10,11,12,13,14,15);
        v16bf af1 = __builtin_shufflevector(a1l, a1h,
            0,1,2,3,4,5,6,7,8,9,10,11,12,13,14,15);
        v16bf bf0 = __builtin_shufflevector(b0l, b0h,
            0,1,2,3,4,5,6,7,8,9,10,11,12,13,14,15);
        v16bf bf1 = __builtin_shufflevector(b1l, b1h,
            0,1,2,3,4,5,6,7,8,9,10,11,12,13,14,15);
        v16bf bf2 = __builtin_shufflevector(b2l, b2h,
            0,1,2,3,4,5,6,7,8,9,10,11,12,13,14,15);
        v16bf bf3 = __builtin_shufflevector(b3l, b3h,
            0,1,2,3,4,5,6,7,8,9,10,11,12,13,14,15);

        // --- 8 WMMAs; each B fragment reused by both M tiles ---
        acc[0][0] = __builtin_amdgcn_wmma_f32_16x16x32_bf16(
            false, af0, false, bf0, (short)0, acc[0][0], false, false);
        acc[1][0] = __builtin_amdgcn_wmma_f32_16x16x32_bf16(
            false, af1, false, bf0, (short)0, acc[1][0], false, false);
        acc[0][1] = __builtin_amdgcn_wmma_f32_16x16x32_bf16(
            false, af0, false, bf1, (short)0, acc[0][1], false, false);
        acc[1][1] = __builtin_amdgcn_wmma_f32_16x16x32_bf16(
            false, af1, false, bf1, (short)0, acc[1][1], false, false);
        acc[0][2] = __builtin_amdgcn_wmma_f32_16x16x32_bf16(
            false, af0, false, bf2, (short)0, acc[0][2], false, false);
        acc[1][2] = __builtin_amdgcn_wmma_f32_16x16x32_bf16(
            false, af1, false, bf2, (short)0, acc[1][2], false, false);
        acc[0][3] = __builtin_amdgcn_wmma_f32_16x16x32_bf16(
            false, af0, false, bf3, (short)0, acc[0][3], false, false);
        acc[1][3] = __builtin_amdgcn_wmma_f32_16x16x32_bf16(
            false, af1, false, bf3, (short)0, acc[1][3], false, false);
    }

    // D: vgpr r -> row base + hi16*8 + r, col l15 (+ 16*nt)
#pragma unroll
    for (int mt = 0; mt < 2; ++mt) {
        const int mbase = mgrp * 32 + mt * 16 + hi16 * 8;
#pragma unroll
        for (int nt = 0; nt < 4; ++nt) {
            const int n    = n0 + nt * 16;
            const float bb = bias[n];
            float* o = out + (size_t)mbase * VOCAB + n;
#pragma unroll
            for (int r = 0; r < 8; ++r) {
                o[(size_t)r * VOCAB] = acc[mt][nt][r] + bb;
            }
        }
    }
}

// ---------------------------------------------------------------------------
// Launch
// ---------------------------------------------------------------------------
extern "C" void kernel_launch(void* const* d_in, const int* in_sizes, int n_in,
                              void* d_out, int out_size, void* d_ws, size_t ws_size,
                              hipStream_t stream) {
    (void)in_sizes; (void)n_in; (void)out_size; (void)ws_size;

    const int*   src     = (const int*)  d_in[0];
    const int*   trg     = (const int*)  d_in[1];
    const float* enc_emb = (const float*)d_in[2];
    const float* enc_Wih = (const float*)d_in[3];
    const float* enc_Whh = (const float*)d_in[4];
    const float* enc_bih = (const float*)d_in[5];
    const float* enc_bhh = (const float*)d_in[6];
    const float* dec_emb = (const float*)d_in[7];
    const float* dec_Wih = (const float*)d_in[8];
    const float* dec_Whh = (const float*)d_in[9];
    const float* dec_bih = (const float*)d_in[10];
    const float* dec_bhh = (const float*)d_in[11];
    const float* fc_W    = (const float*)d_in[12];
    const float* fc_b    = (const float*)d_in[13];
    float* logits = (float*)d_out;

    // workspace carve-up (~21.8 MB total)
    char* ws = (char*)d_ws;
    float*  gi_enc = (float*)ws;  ws += (size_t)BATCH * SRC_T * G3H * 4;
    float*  gi_dec = (float*)ws;  ws += (size_t)BATCH * DEC_T * G3H * 4;
    float*  hT     = (float*)ws;  ws += (size_t)BATCH * HID * 4;
    __bf16* outsb  = (__bf16*)ws; ws += (size_t)M_ROWS * HID * 2;
    __bf16* fcWb   = (__bf16*)ws; ws += (size_t)VOCAB * HID * 2;

    cvt_bf16_kernel<<<(VOCAB * HID + 255) / 256, 256, 0, stream>>>(
        fc_W, fcWb, VOCAB * HID);

    gi_kernel<<<BATCH * SRC_T, G3H, 0, stream>>>(
        src, SRC_T, SRC_T, enc_emb, enc_Wih, enc_bih, gi_enc);

    gru_scan_kernel<<<BATCH, HID, 0, stream>>>(
        gi_enc, nullptr, enc_Whh, enc_bhh, hT, nullptr, SRC_T);

    gi_kernel<<<BATCH * DEC_T, G3H, 0, stream>>>(
        trg, DEC_T, SRC_T, dec_emb, dec_Wih, dec_bih, gi_dec);

    gru_scan_kernel<<<BATCH, HID, 0, stream>>>(
        gi_dec, hT, dec_Whh, dec_bhh, nullptr, outsb, DEC_T);

    // 126 M-groups * 500 N-groups = 63000 waves = 15750 blocks of 4 waves
    fc_gemm_kernel<<<(M_GROUPS * N_GROUPS) / 4, 128, 0, stream>>>(
        outsb, fcWb, fc_b, logits);
}
